// ExponentialLinearAttention_54692113547978
// MI455X (gfx1250) — compile-verified, matrix operand order checked by reading, not run on hardware
//
#include <hip/hip_runtime.h>
#include <hip/hip_bf16.h>
#include <stdint.h>

#define BATCH 8
#define SEQ   4096
#define CH    384
#define NHEAD 8
#define HDIM  48
#define MROWS (BATCH*SEQ)      /* 32768 */
#define QKVG_N (4*CH)          /* 1536  */

typedef __attribute__((ext_vector_type(16))) __bf16 v16bf;
typedef __attribute__((ext_vector_type(8)))  float  v8f;

union FragU { uint4 u[2]; v16bf v; };

#if defined(__AMDGCN__) && __has_builtin(__builtin_amdgcn_tensor_load_to_lds)
#define HAVE_TDM 1
#else
#define HAVE_TDM 0
#endif

__device__ __forceinline__ unsigned short f2bf(float f) {
  unsigned int u = __float_as_uint(f);
  u += 0x7FFFu + ((u >> 16) & 1u);          // round-to-nearest-even
  return (unsigned short)(u >> 16);
}
__device__ __forceinline__ float bf2f(unsigned short s) {
  return __uint_as_float(((unsigned int)s) << 16);
}

// Load a 16-bit WMMA operand fragment: lane (0..15,16..31) -> row (lane&15),
// K chunks at off (VGPR0-3) and off+16 (VGPR4-7), where off = k0 + (lane>>4)*8.
__device__ __forceinline__ v16bf load_frag(const unsigned short* rowp, int off) {
  FragU f;
  f.u[0] = *reinterpret_cast<const uint4*>(rowp + off);
  f.u[1] = *reinterpret_cast<const uint4*>(rowp + off + 16);
  return f.v;
}
__device__ __forceinline__ v16bf load_frag_pad(const unsigned short* rowp,
                                               int off0, int off1, int kmax) {
  FragU f;
  f.u[0] = (off0 < kmax) ? *reinterpret_cast<const uint4*>(rowp + off0) : make_uint4(0,0,0,0);
  f.u[1] = (off1 < kmax) ? *reinterpret_cast<const uint4*>(rowp + off1) : make_uint4(0,0,0,0);
  return f.v;
}

#define WMMA_BF16(a,b,c) \
  __builtin_amdgcn_wmma_f32_16x16x32_bf16(false,(a),false,(b),(short)0,(c),false,false)

#if HAVE_TDM
typedef __attribute__((ext_vector_type(4))) unsigned int tdm_g0_t;
typedef __attribute__((ext_vector_type(8))) int          tdm_g1_t;
typedef __attribute__((ext_vector_type(4))) int          tdm_g2_t;
#if __clang_major__ >= 23
typedef __attribute__((ext_vector_type(8))) int          tdm_g4_t;
#endif

// TDM: DMA a 2D tile (rows x width bf16 elements, row-major, stride = width)
// from global memory into LDS at byte offset lds_off.
__device__ __forceinline__ void tdm_load_2d(unsigned lds_off, const void* gptr,
                                            int width_elems, int rows,
                                            int tensor_rows)
{
  unsigned long long ga = (unsigned long long)(uintptr_t)gptr;
  tdm_g0_t g0;
  g0[0] = 1u;                                              // count=1 (valid), user mode
  g0[1] = lds_off;                                         // lds_addr (bytes)
  g0[2] = (unsigned)(ga & 0xFFFFFFFFu);                    // global_addr[31:0]
  g0[3] = (unsigned)((ga >> 32) & 0x1FFFFFFu) | (2u << 30);// global_addr[56:32] | type=2
  tdm_g1_t g1;
  g1[0] = (int)(1u << 16);                                 // data_size=1 -> 2 bytes
  g1[1] = (int)(((unsigned)width_elems & 0xFFFFu) << 16);  // tensor_dim0[15:0] @ [63:48]
  g1[2] = (int)(((unsigned)width_elems >> 16) |
                (((unsigned)tensor_rows & 0xFFFFu) << 16));// dim0[31:16] | dim1[15:0]
  g1[3] = (int)(((unsigned)tensor_rows >> 16) |
                (((unsigned)width_elems & 0xFFFFu) << 16));// dim1[31:16] | tile_dim0
  g1[4] = rows;                                            // tile_dim1 (tile_dim2=0)
  g1[5] = width_elems;                                     // tensor_dim0_stride[31:0]
  g1[6] = 0;                                               // stride hi / dim1_stride lo
  g1[7] = 0;
  tdm_g2_t gz = {0, 0, 0, 0};
#if __clang_major__ >= 23
  tdm_g4_t gz8 = {0, 0, 0, 0, 0, 0, 0, 0};
  __builtin_amdgcn_tensor_load_to_lds(g0, g1, gz, gz, gz8, 0);
#else
  __builtin_amdgcn_tensor_load_to_lds(g0, g1, gz, gz, 0);
#endif
}
#endif  // HAVE_TDM

// ---------------------------------------------------------------------------
// Generic bf16 GEMM: out[m,n] = sum_k A[m,k]*W[n,k] + bias[n] (+ add[m,n])
// A: [M x K] bf16 row-major (streamed from global, prefetched),
// W: [Nout x K] bf16 row-major (64-row block tile staged into LDS via TDM).
// Block = 256 threads = 8 waves; wave tile 32x32; block tile 128x64. K == CH.
// ---------------------------------------------------------------------------
__global__ __launch_bounds__(256)
void gemm_bf16(const unsigned short* __restrict__ A,
               const unsigned short* __restrict__ W,
               const float* __restrict__ bias,
               const float* __restrict__ addsrc,
               unsigned short* __restrict__ obf,
               float* __restrict__ of32,
               int K, int Nout)
{
  __shared__ unsigned short wsm[64 * CH];                  // 48 KB W tile

  const int lane = threadIdx.x & 31;
  const int wave = threadIdx.x >> 5;
  const int m0 = blockIdx.x * 128 + (wave & 3) * 32;
  const int nblk = blockIdx.y * 64;
  const int n0 = nblk + (wave >> 2) * 32;
  const int r  = lane & 15;
  const int h8 = (lane >> 4) << 3;

  // ---- stage W[nblk .. nblk+63][0..K) into LDS ----
#if HAVE_TDM
  if (threadIdx.x < 32) {                                  // one wave issues the DMA
    tdm_load_2d((unsigned)(uintptr_t)wsm, W + (size_t)nblk * K, K, 64, Nout);
    __builtin_amdgcn_s_wait_tensorcnt(0);
  }
  __syncthreads();
#else
  {
    const uint4* src = (const uint4*)(W + (size_t)nblk * K);
    uint4* dst = (uint4*)wsm;
    for (int i = threadIdx.x; i < 64 * CH / 8; i += 256) dst[i] = src[i];
    __syncthreads();
  }
#endif

  const unsigned short* a0p = A + (size_t)(m0 + r)      * K;
  const unsigned short* a1p = A + (size_t)(m0 + 16 + r) * K;
  const unsigned short* b0p = wsm + (size_t)((wave >> 2) * 32 + r) * K;   // LDS
  const unsigned short* b1p = b0p + (size_t)16 * K;                       // LDS

  v8f acc00 = {}, acc01 = {}, acc10 = {}, acc11 = {};
  for (int k0 = 0; k0 < K; k0 += 32) {
    const int off = k0 + h8;
    if (k0 + 32 < K) {                       // hint next K tile of A into cache
      __builtin_prefetch(a0p + off + 32, 0, 1);
      __builtin_prefetch(a1p + off + 32, 0, 1);
    }
    v16bf a0 = load_frag(a0p, off);          // global_load_b128 x2
    v16bf a1 = load_frag(a1p, off);
    v16bf b0 = load_frag(b0p, off);          // ds_load_b128 x2
    v16bf b1 = load_frag(b1p, off);
    acc00 = WMMA_BF16(a0, b0, acc00);
    acc01 = WMMA_BF16(a0, b1, acc01);
    acc10 = WMMA_BF16(a1, b0, acc10);
    acc11 = WMMA_BF16(a1, b1, acc11);
  }

  v8f accs[2][2] = {{acc00, acc01}, {acc10, acc11}};
  #pragma unroll
  for (int mt = 0; mt < 2; ++mt) {
    #pragma unroll
    for (int nt = 0; nt < 2; ++nt) {
      const int col = n0 + nt * 16 + r;
      const float bv = bias ? bias[col] : 0.0f;
      #pragma unroll
      for (int e = 0; e < 8; ++e) {
        const int row = m0 + mt * 16 + h8 + e;   // C/D layout: lanes16-31 -> M+8
        const size_t idx = (size_t)row * Nout + col;
        float val = accs[mt][nt][e] + bv;
        if (addsrc) val += addsrc[idx];
        if (of32) of32[idx] = val;
        if (obf)  obf[idx]  = f2bf(val);
      }
    }
  }
}

// ---------------------------------------------------------------------------
// Depthwise 3x3 conv over [B, C, 64, 64] stored as [B, N, C]; bf16 output.
// ---------------------------------------------------------------------------
__global__ __launch_bounds__(256)
void dwconv_kernel(const float* __restrict__ x, const float* __restrict__ w,
                   const float* __restrict__ b, unsigned short* __restrict__ t)
{
  const int idx = blockIdx.x * 256 + threadIdx.x;      // < MROWS*CH (exact)
  const int c  = idx % CH;
  const int n  = (idx / CH) & (SEQ - 1);
  const int bb = idx / (CH * SEQ);
  const int y = n >> 6, xx = n & 63;
  float acc = b[c];
  #pragma unroll
  for (int dy = -1; dy <= 1; ++dy) {
    const int yy = y + dy;
    if (yy < 0 || yy > 63) continue;
    #pragma unroll
    for (int dx = -1; dx <= 1; ++dx) {
      const int xc = xx + dx;
      if (xc < 0 || xc > 63) continue;
      acc += x[((size_t)(bb << 12) + (yy << 6) + xc) * CH + c]
           * w[c * 9 + (dy + 1) * 3 + (dx + 1)];
    }
  }
  t[idx] = f2bf(acc);
}

// ---------------------------------------------------------------------------
// phi(q*temp), phi(k), v *= sigmoid(g)  (in-place on bf16 qkvg rows)
// ---------------------------------------------------------------------------
__global__ __launch_bounds__(256)
void phi_gate_kernel(unsigned short* __restrict__ qkvg,
                     const float* __restrict__ temperature)
{
  const int idx = blockIdx.x * 256 + threadIdx.x;      // < MROWS*NHEAD
  const int h = idx & 7;
  const size_t row = (size_t)(idx >> 3);
  unsigned short* base = qkvg + row * QKVG_N;
  unsigned short* q = base + h * HDIM;
  unsigned short* k = base + CH     + h * HDIM;
  unsigned short* v = base + 2 * CH + h * HDIM;
  unsigned short* g = base + 3 * CH + h * HDIM;
  const float tmp = temperature[h];

  float mx = -3.4e38f;
  for (int d = 0; d < HDIM; ++d) mx = fmaxf(mx, bf2f(q[d]) * tmp);
  for (int d = 0; d < HDIM; ++d) q[d] = f2bf(__expf(bf2f(q[d]) * tmp - mx));

  mx = -3.4e38f;
  for (int d = 0; d < HDIM; ++d) mx = fmaxf(mx, bf2f(k[d]));
  for (int d = 0; d < HDIM; ++d) k[d] = f2bf(__expf(bf2f(k[d]) - mx));

  for (int d = 0; d < HDIM; ++d) {
    const float gg = bf2f(g[d]);
    v[d] = f2bf(bf2f(v[d]) / (1.0f + __expf(-gg)));
  }
}

// ---------------------------------------------------------------------------
// kv[e][d] = sum_n k[n,d]*v[n,e], ksum[d] = sum_n k[n,d]; one block per (b,h).
// LDS-staged reduction (~2% of total FLOPs). kv stored TRANSPOSED ([e][d]).
// ---------------------------------------------------------------------------
__global__ __launch_bounds__(256)
void kv_kernel(const unsigned short* __restrict__ qkvg,
               unsigned short* __restrict__ kvt,
               float* __restrict__ ksum)
{
  __shared__ unsigned short kt[32 * HDIM];
  __shared__ unsigned short vt[32 * HDIM];
  const int bh = blockIdx.x;
  const int b = bh >> 3, h = bh & 7;
  const int tid = threadIdx.x;
  float acc[9];
  #pragma unroll
  for (int j = 0; j < 9; ++j) acc[j] = 0.0f;
  float ksacc = 0.0f;
  const int o0 = tid * 9;                    // 256*9 == 48*48 outputs exactly

  for (int nb = 0; nb < SEQ / 32; ++nb) {
    for (int e = tid; e < 32 * HDIM; e += 256) {
      const int nr = e / HDIM, d = e % HDIM;
      const size_t rowoff = (size_t)((b << 12) + (nb << 5) + nr) * QKVG_N;
      kt[e] = qkvg[rowoff + CH     + h * HDIM + d];
      vt[e] = qkvg[rowoff + 2 * CH + h * HDIM + d];
    }
    __syncthreads();
    #pragma unroll
    for (int j = 0; j < 9; ++j) {
      const int o = o0 + j;
      const int d = o / HDIM, ee = o % HDIM;
      float s = 0.0f;
      for (int i = 0; i < 32; ++i)
        s += bf2f(kt[i * HDIM + d]) * bf2f(vt[i * HDIM + ee]);
      acc[j] += s;
    }
    if (tid < HDIM)
      for (int i = 0; i < 32; ++i) ksacc += bf2f(kt[i * HDIM + tid]);
    __syncthreads();
  }
  #pragma unroll
  for (int j = 0; j < 9; ++j) {
    const int o = o0 + j;
    const int d = o / HDIM, ee = o % HDIM;
    kvt[(size_t)bh * (HDIM * HDIM) + ee * HDIM + d] = f2bf(acc[j]);
  }
  if (tid < HDIM) ksum[(size_t)bh * HDIM + tid] = ksacc;
}

// den[row,h] = dot(q_row, ksum) + EPS
__global__ __launch_bounds__(256)
void den_kernel(const unsigned short* __restrict__ qkvg,
                const float* __restrict__ ksum,
                float* __restrict__ den)
{
  const int idx = blockIdx.x * 256 + threadIdx.x;      // < MROWS*NHEAD
  const int h = idx & 7;
  const size_t row = (size_t)(idx >> 3);
  const int b = (int)(row >> 12);
  const unsigned short* q = qkvg + row * QKVG_N + h * HDIM;
  const float* s = ksum + (size_t)((b << 3) + h) * HDIM;
  float acc = 1e-6f;
  for (int d = 0; d < HDIM; ++d) acc += bf2f(q[d]) * s[d];
  den[idx] = acc;
}

// ---------------------------------------------------------------------------
// attn[row, h*48 + e] = (q_row . kv[:,e]) / den[row,h]   via WMMA, K=48 padded
// ---------------------------------------------------------------------------
__global__ __launch_bounds__(256)
void num_kernel(const unsigned short* __restrict__ qkvg,
                const unsigned short* __restrict__ kvt,
                const float* __restrict__ den,
                unsigned short* __restrict__ attn)
{
  const int lane = threadIdx.x & 31;
  const int gw = blockIdx.x * 8 + (threadIdx.x >> 5);
  const int h  = gw & 7;
  const int rt = gw >> 3;                    // 0..2047
  const int m0 = rt << 4;
  const int b  = m0 >> 12;
  const int r  = lane & 15;
  const int h8 = (lane >> 4) << 3;

  const unsigned short* arow   = qkvg + (size_t)(m0 + r) * QKVG_N + h * HDIM;
  const unsigned short* kvbase = kvt + (size_t)((b << 3) + h) * (HDIM * HDIM);

  v8f acc[3] = {};
  #pragma unroll
  for (int ks = 0; ks < 2; ++ks) {
    const int off0 = ks * 32 + h8;
    const int off1 = off0 + 16;
    v16bf a = load_frag_pad(arow, off0, off1, HDIM);
    #pragma unroll
    for (int nt = 0; nt < 3; ++nt) {
      const unsigned short* brow = kvbase + (size_t)(nt * 16 + r) * HDIM;
      v16bf bb = load_frag_pad(brow, off0, off1, HDIM);
      acc[nt] = WMMA_BF16(a, bb, acc[nt]);
    }
  }

  float inv[8];
  #pragma unroll
  for (int e = 0; e < 8; ++e) {
    const int row = m0 + h8 + e;
    inv[e] = 1.0f / den[(size_t)row * NHEAD + h];
  }
  #pragma unroll
  for (int nt = 0; nt < 3; ++nt) {
    const int col = h * HDIM + nt * 16 + r;
    #pragma unroll
    for (int e = 0; e < 8; ++e) {
      const int row = m0 + h8 + e;
      attn[(size_t)row * CH + col] = f2bf(acc[nt][e] * inv[e]);
    }
  }
}

// ---------------------------------------------------------------------------
__global__ __launch_bounds__(256)
void pack_bf16_kernel(const float* __restrict__ src,
                      unsigned short* __restrict__ dst, int n)
{
  const int i = blockIdx.x * 256 + threadIdx.x;
  if (i < n) dst[i] = f2bf(src[i]);
}
__global__ __launch_bounds__(256)
void copy_f32_kernel(const float* __restrict__ src,
                     float* __restrict__ dst, int n)
{
  const int i = blockIdx.x * 256 + threadIdx.x;
  if (i < n) dst[i] = src[i];
}

// ---------------------------------------------------------------------------
extern "C" void kernel_launch(void* const* d_in, const int* in_sizes, int n_in,
                              void* d_out, int out_size, void* d_ws, size_t ws_size,
                              hipStream_t stream)
{
  const float* x    = (const float*)d_in[0];
  const float* wq   = (const float*)d_in[1];
  const float* bq   = (const float*)d_in[2];
  const float* wk   = (const float*)d_in[3];
  const float* bk   = (const float*)d_in[4];
  const float* wv   = (const float*)d_in[5];
  const float* bv   = (const float*)d_in[6];
  const float* wg   = (const float*)d_in[7];
  const float* bg   = (const float*)d_in[8];
  const float* wo   = (const float*)d_in[9];
  const float* bo   = (const float*)d_in[10];
  const float* temp = (const float*)d_in[11];
  const float* dw_w = (const float*)d_in[12];
  const float* dw_b = (const float*)d_in[13];
  const float* pw_w = (const float*)d_in[14];
  const float* pw_b = (const float*)d_in[15];
  float* out = (float*)d_out;

  char* ws = (char*)d_ws;
  unsigned short* t_bf  = (unsigned short*)(ws);              // M*C bf16
  unsigned short* xa_bf = (unsigned short*)(ws +  25165824);  // M*C bf16
  unsigned short* qkvg  = (unsigned short*)(ws +  50331648);  // M*1536 bf16
  unsigned short* attn  = (unsigned short*)(ws + 150994944);  // M*C bf16
  unsigned short* pw_bf = (unsigned short*)(ws + 176160768);  // 384*384 bf16
  unsigned short* wcat  = (unsigned short*)(ws + 176455680);  // 1536*384 bf16
  unsigned short* wo_bf = (unsigned short*)(ws + 177635328);  // 384*384 bf16
  float*          bcat  = (float*)        (ws + 177930240);   // 1536 f32
  unsigned short* kvt   = (unsigned short*)(ws + 177936384);  // 64*48*48 bf16
  float*          ksum  = (float*)        (ws + 178231296);   // 64*48 f32
  float*          den   = (float*)        (ws + 178243584);   // M*8 f32
  // total ws used: 179292160 bytes (~171 MB)

  const int WE = CH * CH;                                     // 147456
  dim3 blk(256);

  pack_bf16_kernel<<<(WE + 255) / 256, blk, 0, stream>>>(pw_w, pw_bf, WE);
  pack_bf16_kernel<<<(WE + 255) / 256, blk, 0, stream>>>(wq, wcat + 0 * WE, WE);
  pack_bf16_kernel<<<(WE + 255) / 256, blk, 0, stream>>>(wk, wcat + 1 * WE, WE);
  pack_bf16_kernel<<<(WE + 255) / 256, blk, 0, stream>>>(wv, wcat + 2 * WE, WE);
  pack_bf16_kernel<<<(WE + 255) / 256, blk, 0, stream>>>(wg, wcat + 3 * WE, WE);
  pack_bf16_kernel<<<(WE + 255) / 256, blk, 0, stream>>>(wo, wo_bf, WE);
  copy_f32_kernel<<<2, blk, 0, stream>>>(bq, bcat + 0 * CH, CH);
  copy_f32_kernel<<<2, blk, 0, stream>>>(bk, bcat + 1 * CH, CH);
  copy_f32_kernel<<<2, blk, 0, stream>>>(bv, bcat + 2 * CH, CH);
  copy_f32_kernel<<<2, blk, 0, stream>>>(bg, bcat + 3 * CH, CH);

  // t = depthwise3x3(x) + dw_b                         (bf16)
  dwconv_kernel<<<(MROWS * CH) / 256, blk, 0, stream>>>(x, dw_w, dw_b, t_bf);

  // xa = x + t @ pw_w^T + pw_b                         (WMMA + TDM)
  gemm_bf16<<<dim3(MROWS / 128, CH / 64), blk, 0, stream>>>(
      t_bf, pw_bf, pw_b, x, xa_bf, nullptr, CH, CH);

  // qkvg = xa @ [wq;wk;wv;wg]^T + [bq;bk;bv;bg]        (WMMA + TDM)
  gemm_bf16<<<dim3(MROWS / 128, QKVG_N / 64), blk, 0, stream>>>(
      xa_bf, wcat, bcat, nullptr, qkvg, nullptr, CH, QKVG_N);

  // q = phi(q*temp), k = phi(k), v *= sigmoid(g)       (in-place)
  phi_gate_kernel<<<(MROWS * NHEAD) / 256, blk, 0, stream>>>(qkvg, temp);

  // kv^T and ksum per (b,h)                            (LDS reduction)
  kv_kernel<<<BATCH * NHEAD, blk, 0, stream>>>(qkvg, kvt, ksum);

  // den = q . ksum + eps
  den_kernel<<<(MROWS * NHEAD) / 256, blk, 0, stream>>>(qkvg, ksum, den);

  // attn = (q @ kv) / den                              (WMMA, K=48 padded)
  num_kernel<<<(MROWS / 16) * NHEAD / 8, blk, 0, stream>>>(qkvg, kvt, den, attn);

  // out = attn @ wo^T + bo                             (f32 out, WMMA + TDM)
  gemm_bf16<<<dim3(MROWS / 128, CH / 64), blk, 0, stream>>>(
      attn, wo_bf, bo, nullptr, nullptr, out, CH, CH);
}
